// FieldFactorizeLayer_56813827391597
// MI455X (gfx1250) — matrix operand order repaired and checked by reading, not compiled, not run on hardware
//
#include <hip/hip_runtime.h>

// ---------------------------------------------------------------------------
// FFM-style pairwise field interaction, MI455X (gfx1250, wave32).
// Gather-bandwidth bound (~780 MB/call -> ~33us @ 23.3 TB/s). Pooling via
// coalesced per-row float2 gathers; pair dot-products via v_wmma_f32_16x16x32_f16
// (diagonal of X * Y^T accumulated across pairs).
// ---------------------------------------------------------------------------

#define FNUM   8
#define NSEQ   2
#define BATCH  4096
#define SEQLEN 50
#define VOCAB  100000
#define EMB    64

#define BT      16            // batch tile per block (one WMMA M/N)
#define NPAIRS  28
#define PSPLIT  4             // pair groups across grid.y
#define PITCH   80            // halfs per pooled row: 160B, 32B-aligned rows

typedef _Float16 v16h __attribute__((ext_vector_type(16)));
typedef _Float16 h8   __attribute__((ext_vector_type(8)));
typedef _Float16 h2   __attribute__((ext_vector_type(2)));
typedef float    v8f  __attribute__((ext_vector_type(8)));

__constant__ unsigned char PAIR_A[NPAIRS] = {0,0,0,0,0,0,0, 1,1,1,1,1,1, 2,2,2,2,2, 3,3,3,3, 4,4,4, 5,5, 6};
__constant__ unsigned char PAIR_T[NPAIRS] = {1,2,3,4,5,6,7, 2,3,4,5,6,7, 3,4,5,6,7, 4,5,6,7, 5,6,7, 6,7, 7};

// Pool two rows (batch items) of one table: per-lane float2 covers e=2*lane..2*lane+1.
// Two independent accumulation chains, unrolled for memory-level parallelism.
__device__ __forceinline__ void pool_two(const float* __restrict__ base,
                                         const int* __restrict__ idx0,
                                         const int* __restrict__ idx1,
                                         int La, int lane,
                                         float2& s0, float2& s1) {
  s0 = make_float2(0.f, 0.f);
  s1 = make_float2(0.f, 0.f);
#pragma unroll 5
  for (int l = 0; l < La; ++l) {
    const size_t v0 = (size_t)idx0[l] * EMB;
    const size_t v1 = (size_t)idx1[l] * EMB;
    const float2 r0 = *(const float2*)(base + v0 + 2 * lane);
    const float2 r1 = *(const float2*)(base + v1 + 2 * lane);
    s0.x += r0.x; s0.y += r0.y;
    s1.x += r1.x; s1.y += r1.y;
  }
}

extern "C" __global__ void __launch_bounds__(256)
ffm_pair_kernel(const int* __restrict__ x_seq,     // [NSEQ][BATCH][SEQLEN]
                const int* __restrict__ x_cat,     // [FNUM-NSEQ][BATCH]
                const float* __restrict__ tables,  // [FNUM][FNUM][VOCAB][EMB]
                float* __restrict__ dst,           // [psplit][BATCH]
                int psplit, int ppg) {
  __shared__ int s_seq[NSEQ][BT][SEQLEN];
  __shared__ int s_cat[FNUM - NSEQ][BT];
  __shared__ __align__(64) _Float16 sX[BT][PITCH];
  __shared__ __align__(64) _Float16 sY[BT][PITCH];
  __shared__ float sDiag[2][BT];

  const int tid  = threadIdx.x;
  const int wave = tid >> 5;
  const int lane = tid & 31;
  const int b0   = blockIdx.x * BT;
  const int pg   = blockIdx.y;

  // ---- stage indices for this batch tile into LDS ----
  for (int i = tid; i < NSEQ * BT * SEQLEN; i += 256) {
    const int a = i / (BT * SEQLEN);
    const int r = i % (BT * SEQLEN);
    const int b = r / SEQLEN;
    const int l = r % SEQLEN;
    s_seq[a][b][l] = x_seq[((size_t)a * BATCH + b0 + b) * SEQLEN + l];
  }
  if (tid < (FNUM - NSEQ) * BT) {
    const int c = tid / BT, b = tid % BT;
    s_cat[c][b] = x_cat[(size_t)c * BATCH + b0 + b];
  }
  __syncthreads();

  v8f acc = {};  // C tile (waves 0 and 1 only; e-chunks 0..31 / 32..63)

  for (int pp = 0; pp < ppg; ++pp) {
    const int p  = pp * psplit + pg;     // stripe pairs across grid.y for balance
    const int fa = PAIR_A[p];
    const int ft = PAIR_T[p];

    // Wave w pools X rows {w, w+8} (field fa via table fa->ft)
    //           and Y rows {w, w+8} (field ft via table ft->fa).
    const float* baseA = tables + (size_t)(fa * FNUM + ft) * VOCAB * EMB;
    const float* baseB = tables + (size_t)(ft * FNUM + fa) * VOCAB * EMB;
    const int LaA = (fa < NSEQ) ? SEQLEN : 1;
    const int LaB = (ft < NSEQ) ? SEQLEN : 1;
    const int* ia0 = (fa < NSEQ) ? &s_seq[fa][wave][0]     : &s_cat[fa - NSEQ][wave];
    const int* ia1 = (fa < NSEQ) ? &s_seq[fa][wave + 8][0] : &s_cat[fa - NSEQ][wave + 8];
    const int* ib0 = (ft < NSEQ) ? &s_seq[ft][wave][0]     : &s_cat[ft - NSEQ][wave];
    const int* ib1 = (ft < NSEQ) ? &s_seq[ft][wave + 8][0] : &s_cat[ft - NSEQ][wave + 8];

    float2 xa0, xa1, yb0, yb1;
    pool_two(baseA, ia0, ia1, LaA, lane, xa0, xa1);   // overlaps prev pair's WMMA
    pool_two(baseB, ib0, ib1, LaB, lane, yb0, yb1);
    const float sA = 1.0f / (float)LaA;
    const float sB = 1.0f / (float)LaB;

    __syncthreads();  // prev pair's WMMA reads of sX/sY are done

    *(h2*)&sX[wave][2 * lane]     = (h2){(_Float16)(xa0.x * sA), (_Float16)(xa0.y * sA)};
    *(h2*)&sX[wave + 8][2 * lane] = (h2){(_Float16)(xa1.x * sA), (_Float16)(xa1.y * sA)};
    *(h2*)&sY[wave][2 * lane]     = (h2){(_Float16)(yb0.x * sB), (_Float16)(yb0.y * sB)};
    *(h2*)&sY[wave + 8][2 * lane] = (h2){(_Float16)(yb1.x * sB), (_Float16)(yb1.y * sB)};

    __syncthreads();  // pooled tiles visible

    if (wave < 2) {   // wave-uniform branch: EXEC all-ones inside (WMMA legal)
      const int k0 = wave * 32;     // K-chunk of E
      const int m  = lane & 15;
      const int hi = lane >> 4;
      // A fragment (16-bit A 16x32 layout, ISA 7.12.2): lane m holds
      // K = k0 + {8*hi..8*hi+7} in v0..v3 halves, K = k0+16+{8*hi..} in v4..v7.
      const _Float16* xr = &sX[m][0];
      const h8 alo = *(const h8*)(xr + k0 + 8 * hi);
      const h8 ahi = *(const h8*)(xr + k0 + 16 + 8 * hi);
      // B fragment (32x16, K x N with N=lane&15): lane holds K = k0+{8*hi..8*hi+7}
      // then K = k0+16+{8*hi..} of column n, from row n of Y (since B[k][n] = Y[n][k]).
      const _Float16* yr = &sY[m][0];
      const h8 blo = *(const h8*)(yr + k0 + 8 * hi);
      const h8 bhi = *(const h8*)(yr + k0 + 16 + 8 * hi);
      v16h A, Bm;
#pragma unroll
      for (int i = 0; i < 8; ++i) {
        A[i] = alo[i];  A[i + 8] = ahi[i];
        Bm[i] = blo[i]; Bm[i + 8] = bhi[i];
      }
      // C[m][n] += sum_e X[m][e] * Y[n][e]; diagonal is the pair contribution.
      acc = __builtin_amdgcn_wmma_f32_16x16x32_f16(
          /*neg_a=*/false, A, /*neg_b=*/false, Bm,
          /*c_mod=*/(short)0, acc, /*reuse_a=*/false, /*reuse_b=*/false);
    }
  }

  // ---- extract diagonal C[b][b] and reduce across the two K-chunk waves ----
  if (wave < 2) {
    const int j = lane & 7;
    float d;
    switch (j) {  // static vector indexing only
      case 0: d = acc[0]; break; case 1: d = acc[1]; break;
      case 2: d = acc[2]; break; case 3: d = acc[3]; break;
      case 4: d = acc[4]; break; case 5: d = acc[5]; break;
      case 6: d = acc[6]; break; default: d = acc[7]; break;
    }
    // C layout: lanes 0-15 -> N=lane, M=vgpr; lanes 16-31 -> N=lane-16, M=vgpr+8.
    // Diagonal b<8: lane b, vgpr b. Diagonal b>=8: lane b+16, vgpr b-8.
    if (lane < 8)        sDiag[wave][lane]      = d;
    else if (lane >= 24) sDiag[wave][lane - 16] = d;
  }
  __syncthreads();

  if (tid < BT) {
    dst[(size_t)pg * BATCH + b0 + tid] = sDiag[0][tid] + sDiag[1][tid];
  }
}

extern "C" __global__ void ffm_reduce_kernel(const float* __restrict__ partial,
                                             float* __restrict__ out, int ng) {
  const int i = blockIdx.x * blockDim.x + threadIdx.x;
  if (i < BATCH) {
    float s = 0.f;
    for (int g = 0; g < ng; ++g) s += partial[(size_t)g * BATCH + i];
    out[i] = s;
  }
}

extern "C" void kernel_launch(void* const* d_in, const int* in_sizes, int n_in,
                              void* d_out, int out_size, void* d_ws, size_t ws_size,
                              hipStream_t stream) {
  const int*   x_seq  = (const int*)d_in[0];   // (2, 4096, 50) int32
  const int*   x_cat  = (const int*)d_in[1];   // (6, 4096, 1)  int32
  const float* tables = (const float*)d_in[2]; // (8, 8, 100000, 64) f32
  float*       out    = (float*)d_out;         // (4096, 1, 1)  f32

  const size_t need = (size_t)PSPLIT * BATCH * sizeof(float);
  if (d_ws != nullptr && ws_size >= need) {
    // Pair-split path: 1024 blocks for occupancy, partials in d_ws, then reduce.
    dim3 grid(BATCH / BT, PSPLIT);
    ffm_pair_kernel<<<grid, 256, 0, stream>>>(x_seq, x_cat, tables,
                                              (float*)d_ws, PSPLIT, NPAIRS / PSPLIT);
    ffm_reduce_kernel<<<(BATCH + 255) / 256, 256, 0, stream>>>((const float*)d_ws,
                                                               out, PSPLIT);
  } else {
    // Fallback: single pair-group writes final sums directly to d_out.
    dim3 grid(BATCH / BT, 1);
    ffm_pair_kernel<<<grid, 256, 0, stream>>>(x_seq, x_cat, tables, out, 1, NPAIRS);
  }
}
